// MultiHeadAttention_11562051961004
// MI455X (gfx1250) — matrix operand order repaired and checked by reading, not compiled, not run on hardware
//
#include <hip/hip_runtime.h>
#include <hip/hip_bf16.h>
#include <stdint.h>

// ---------------------------------------------------------------------------
// MI455X (gfx1250) multi-head attention forward.
//   qkv = x @ w_qkv^T                      (WMMA bf16, fp32 accum)
//   scores = 0.125 * q @ k^T  per (b,h)    (WMMA bf16) -> d_out attn region
//   softmax rows (in place, fp32)
//   ctx = attn @ v            per (b,h)    (WMMA bf16, B staged N-major)
//   out = ctx @ w_proj^T + b               (WMMA bf16)
// Software-pipelined K loop: global loads for tile k+1 are issued between the
// LDS commit and the WMMAs of tile k, hiding HBM/L2 latency behind matrix math.
// Workspace layout (needs ~101 MB): qkv fp32 [8192,2304], ctx fp32 [8192,768]
// ---------------------------------------------------------------------------

typedef __bf16 bf16_t;
typedef __attribute__((ext_vector_type(16))) __bf16 v16bf;
typedef __attribute__((ext_vector_type(8)))  __bf16 v8bf;
typedef __attribute__((ext_vector_type(8)))  float  v8f;
typedef __attribute__((ext_vector_type(8)))  float  v8ff;

#define BM 128
#define BN 64
#define BK 32
#define LDS_K 40   // 32 + 8 bf16 pad per row (80B row stride, 16B aligned)

// convert 8 fp32 -> 8 bf16 (v_cvt_pk_bf16_f32 x4) and store as one
// 16B LDS write (ds_store_b128)
__device__ __forceinline__ void store8_bf16(bf16_t* dst, float4 a, float4 b) {
  v8ff f = {a.x, a.y, a.z, a.w, b.x, b.y, b.z, b.w};
  union { v8bf v; uint4 u; } cv;
  cv.v = __builtin_convertvector(f, v8bf);
  *reinterpret_cast<uint4*>(dst) = cv.u;
}

union FragBF16 { uint4 u4[2]; v16bf v; };

// A-matrix 16x32 bf16 fragment (ISA 7.12.2): lane row = lane&15, h = lane>>4,
// elements 0..7 = K[8h..8h+7], elements 8..15 = K[8h+16..8h+23]
__device__ __forceinline__ v16bf load_frag_a(const bf16_t* rowptr, int h) {
  FragBF16 f;
  f.u4[0] = *reinterpret_cast<const uint4*>(rowptr + 8 * h);
  f.u4[1] = *reinterpret_cast<const uint4*>(rowptr + 8 * h + 16);
  return f.v;
}

// B-matrix 32x16 bf16 fragment: lane col = lane&15, h = lane>>4,
// elements 0..7 = K[8h..8h+7], elements 8..15 = K[8h+16..8h+23]
__device__ __forceinline__ v16bf load_frag_b(const bf16_t* colptr, int h) {
  FragBF16 f;
  f.u4[0] = *reinterpret_cast<const uint4*>(colptr + 8 * h);
  f.u4[1] = *reinterpret_cast<const uint4*>(colptr + 8 * h + 16);
  return f.v;
}

// C = alpha * A @ B(^T) [+ bias], fp32 in/out, bf16 WMMA compute.
//   A: [M,K]  row-major, leading dim lda (K contiguous)
//   B: if !B_N_MAJOR: [Ncols,K] row-major (ldb), i.e. B^T, K contiguous
//      if  B_N_MAJOR: [K,Ncols] row-major (ldb), N contiguous (LDS-transposed)
//   batching: z = blockIdx.z -> (zb, zh) = (z / Hdiv, z % Hdiv),
//             each operand offset by zb*sb + zh*sh elements.
// Requires: M % 128 == 0, Ncols % 64 == 0, K % 32 == 0, 16B-aligned pointers.
template <bool B_N_MAJOR, bool HAS_BIAS>
__global__ __launch_bounds__(256)
void gemm_wmma_bf16(const float* __restrict__ A, const float* __restrict__ B,
                    float* __restrict__ C, const float* __restrict__ bias,
                    int M, int Ncols, int K,
                    long lda, long ldb, long ldc,
                    int Hdiv,
                    long a_sb, long a_sh, long b_sb, long b_sh,
                    long c_sb, long c_sh,
                    float alpha) {
  __shared__ __align__(16) bf16_t As[BM][LDS_K];
  __shared__ __align__(16) bf16_t Bs[BN][LDS_K];

  const int z  = blockIdx.z;
  const int zb = z / Hdiv;
  const int zh = z - zb * Hdiv;
  const float* Az = A + (long)zb * a_sb + (long)zh * a_sh;
  const float* Bz = B + (long)zb * b_sb + (long)zh * b_sh;
  float*       Cz = C + (long)zb * c_sb + (long)zh * c_sh;

  const long blockM = (long)blockIdx.y * BM;
  const long blockN = (long)blockIdx.x * BN;

  const int t    = threadIdx.x;
  const int lane = t & 31;
  const int wid  = t >> 5;            // 8 wave32s
  const int wm   = (wid >> 1) * 32;   // wave M offset: 0/32/64/96
  const int wn   = (wid & 1) * 32;    // wave N offset: 0/32
  const int lrow = lane & 15;
  const int lh   = lane >> 4;

  // Per-thread staging coordinates (two A chunks, one B chunk per K tile).
  const int arow0 = t >> 2;            // rows 0..63
  const int arow1 = arow0 + 64;        // rows 64..127
  const int akc   = (t & 3) * 8;       // k sub-offset 0/8/16/24
  const float* aptr0 = Az + (blockM + arow0) * lda + akc;
  const float* aptr1 = Az + (blockM + arow1) * lda + akc;

  const int bnrow = t >> 2;            // K-major: 64 n-rows x 4 k-chunks
  const int bkk   = t >> 3;            // N-major: 32 k-rows x 8 n-chunks
  const int bn0   = (t & 7) * 8;
  const float* bptr = B_N_MAJOR ? (Bz + (long)bkk * ldb + (blockN + bn0))
                                : (Bz + (blockN + bnrow) * ldb + akc);

  v8f acc[2][2] = {{{}, {}}, {{}, {}}};
  float4 af[4], bfr[2];

  // Prologue: fetch K-tile 0 into registers.
  af[0]  = *reinterpret_cast<const float4*>(aptr0);
  af[1]  = *reinterpret_cast<const float4*>(aptr0 + 4);
  af[2]  = *reinterpret_cast<const float4*>(aptr1);
  af[3]  = *reinterpret_cast<const float4*>(aptr1 + 4);
  bfr[0] = *reinterpret_cast<const float4*>(bptr);
  bfr[1] = *reinterpret_cast<const float4*>(bptr + 4);

  for (int k0 = 0; k0 < K; k0 += BK) {
    // ---- commit staged registers to LDS (fp32 -> bf16) ----
    store8_bf16(&As[arow0][akc], af[0], af[1]);
    store8_bf16(&As[arow1][akc], af[2], af[3]);
    if (!B_N_MAJOR) {
      store8_bf16(&Bs[bnrow][akc], bfr[0], bfr[1]);
    } else {  // transpose [k][n] -> LDS [n][k]
      Bs[bn0 + 0][bkk] = (bf16_t)bfr[0].x;  Bs[bn0 + 1][bkk] = (bf16_t)bfr[0].y;
      Bs[bn0 + 2][bkk] = (bf16_t)bfr[0].z;  Bs[bn0 + 3][bkk] = (bf16_t)bfr[0].w;
      Bs[bn0 + 4][bkk] = (bf16_t)bfr[1].x;  Bs[bn0 + 5][bkk] = (bf16_t)bfr[1].y;
      Bs[bn0 + 6][bkk] = (bf16_t)bfr[1].z;  Bs[bn0 + 7][bkk] = (bf16_t)bfr[1].w;
    }
    __syncthreads();

    // ---- issue next K-tile's global loads; they complete under the WMMAs ----
    if (k0 + BK < K) {
      const long ko = k0 + BK;
      af[0] = *reinterpret_cast<const float4*>(aptr0 + ko);
      af[1] = *reinterpret_cast<const float4*>(aptr0 + ko + 4);
      af[2] = *reinterpret_cast<const float4*>(aptr1 + ko);
      af[3] = *reinterpret_cast<const float4*>(aptr1 + ko + 4);
      const float* bp = B_N_MAJOR ? bptr + ko * ldb : bptr + ko;
      bfr[0] = *reinterpret_cast<const float4*>(bp);
      bfr[1] = *reinterpret_cast<const float4*>(bp + 4);
    }

    // ---- 4 WMMAs per wave: 32x32 patch, K=32 step ----
    v16bf a0 = load_frag_a(&As[wm + lrow][0], lh);
    v16bf a1 = load_frag_a(&As[wm + 16 + lrow][0], lh);
    v16bf b0 = load_frag_b(&Bs[wn + lrow][0], lh);
    v16bf b1 = load_frag_b(&Bs[wn + 16 + lrow][0], lh);
    acc[0][0] = __builtin_amdgcn_wmma_f32_16x16x32_bf16(
        false, a0, false, b0, (short)0, acc[0][0], false, false);
    acc[0][1] = __builtin_amdgcn_wmma_f32_16x16x32_bf16(
        false, a0, false, b1, (short)0, acc[0][1], false, false);
    acc[1][0] = __builtin_amdgcn_wmma_f32_16x16x32_bf16(
        false, a1, false, b0, (short)0, acc[1][0], false, false);
    acc[1][1] = __builtin_amdgcn_wmma_f32_16x16x32_bf16(
        false, a1, false, b1, (short)0, acc[1][1], false, false);
    __syncthreads();
  }

  // ---- epilogue: C/D layout — VGPR r: M = r + 8*(lane>>4), N = lane&15 ----
  #pragma unroll
  for (int i = 0; i < 2; ++i) {
    #pragma unroll
    for (int j = 0; j < 2; ++j) {
      const long row0 = blockM + wm + 16 * i + 8 * lh;
      const long col  = blockN + wn + 16 * j + lrow;
      const float b = HAS_BIAS ? bias[col] : 0.0f;
      #pragma unroll
      for (int r = 0; r < 8; ++r) {
        Cz[(row0 + r) * ldc + col] = acc[i][j][r] * alpha + b;
      }
    }
  }
}

// In-place softmax over rows of length 1024 (one 256-thread block per row).
__global__ __launch_bounds__(256)
void softmax_rows_1024(float* __restrict__ attn) {
  __shared__ float red[256];
  const long row = blockIdx.x;
  float* p = attn + row * 1024;
  const int t = threadIdx.x;

  float4 v = *reinterpret_cast<float4*>(&p[t * 4]);
  float m = fmaxf(fmaxf(v.x, v.y), fmaxf(v.z, v.w));
  red[t] = m;
  __syncthreads();
  for (int s = 128; s > 0; s >>= 1) {
    if (t < s) red[t] = fmaxf(red[t], red[t + s]);
    __syncthreads();
  }
  const float rowmax = red[0];
  __syncthreads();

  v.x = __expf(v.x - rowmax); v.y = __expf(v.y - rowmax);
  v.z = __expf(v.z - rowmax); v.w = __expf(v.w - rowmax);
  red[t] = v.x + v.y + v.z + v.w;
  __syncthreads();
  for (int s = 128; s > 0; s >>= 1) {
    if (t < s) red[t] += red[t + s];
    __syncthreads();
  }
  const float inv = 1.0f / red[0];
  v.x *= inv; v.y *= inv; v.z *= inv; v.w *= inv;
  *reinterpret_cast<float4*>(&p[t * 4]) = v;
}

extern "C" void kernel_launch(void* const* d_in, const int* in_sizes, int n_in,
                              void* d_out, int out_size, void* d_ws,
                              size_t ws_size, hipStream_t stream) {
  (void)in_sizes; (void)n_in; (void)out_size; (void)ws_size;

  const float* x      = (const float*)d_in[0];  // [8, 1024, 768]
  const float* w_qkv  = (const float*)d_in[1];  // [2304, 768]
  const float* w_proj = (const float*)d_in[2];  // [768, 768]
  const float* b_proj = (const float*)d_in[3];  // [768]

  constexpr long Bb = 8, Nn = 1024, Cc = 768, Hh = 12, Dd = 64;
  constexpr long Rows = Bb * Nn;                // 8192

  float* out  = (float*)d_out;                  // [8192, 768]
  float* attn = out + Bb * Nn * Cc;             // [8,12,1024,1024]
  float* qkv  = (float*)d_ws;                   // [8192, 2304] fp32
  float* ctx  = qkv + Rows * 3 * Cc;            // [8192, 768]  fp32

  const dim3 blk(256);

  // 1) qkv = x @ w_qkv^T   (x L2-resident across N-tile re-reads: 25MB << 192MB L2)
  gemm_wmma_bf16<false, false><<<dim3(3 * Cc / BN, Rows / BM, 1), blk, 0, stream>>>(
      x, w_qkv, qkv, nullptr,
      (int)Rows, (int)(3 * Cc), (int)Cc,
      Cc, Cc, 3 * Cc,
      1, 0, 0, 0, 0, 0, 0, 1.0f);

  // 2) scores = 0.125 * q @ k^T per (b,h)  -> attn region of d_out
  gemm_wmma_bf16<false, false><<<dim3(Nn / BN, Nn / BM, Bb * Hh), blk, 0, stream>>>(
      qkv, qkv + Cc, attn, nullptr,
      (int)Nn, (int)Nn, (int)Dd,
      3 * Cc, 3 * Cc, Nn,
      (int)Hh, Nn * 3 * Cc, Dd, Nn * 3 * Cc, Dd, Hh * Nn * Nn, Nn * Nn,
      0.125f);

  // 3) softmax in place over 98304 rows of 1024
  softmax_rows_1024<<<dim3((unsigned)(Bb * Hh * Nn)), blk, 0, stream>>>(attn);

  // 4) ctx = attn @ v per (b,h); v is [m,d] row-major -> N-major B staging
  gemm_wmma_bf16<true, false><<<dim3(Dd / BN, Nn / BM, Bb * Hh), blk, 0, stream>>>(
      attn, qkv + 2 * Cc, ctx, nullptr,
      (int)Nn, (int)Dd, (int)Nn,
      Nn, 3 * Cc, Cc,
      (int)Hh, Hh * Nn * Nn, Nn * Nn, Nn * 3 * Cc, Dd, Nn * Cc, Dd,
      1.0f);

  // 5) out = ctx @ w_proj^T + b_proj
  gemm_wmma_bf16<false, true><<<dim3(Cc / BN, Rows / BM, 1), blk, 0, stream>>>(
      ctx, w_proj, out, b_proj,
      (int)Rows, (int)Cc, (int)Cc,
      Cc, Cc, Cc,
      1, 0, 0, 0, 0, 0, 0, 1.0f);
}
